// GAT_78408922956188
// MI455X (gfx1250) — compile-verified, matrix operand order checked by reading, not run on hardware
//
#include <hip/hip_runtime.h>
#include <hip/hip_bf16.h>

#define N_NODES 131072
#define N_EDGES 1048576
#define NE_TOT  (N_EDGES + N_NODES)   // self-loops appended
#define F_IN    128
#define INT_D   256
#define HEADS1  8
#define HC1     64                    // 8 heads * 8 ch
#define C2      256
#define NEG_SLOPE 0.2f
#define BN_EPS  1e-5f
#define ORD_NEG_INF 0x007FFFFFu       // ordered-int encoding of -inf

typedef __attribute__((ext_vector_type(16))) __bf16 v16bf;
typedef __attribute__((ext_vector_type(8)))  float  v8f;
typedef unsigned short ushortT;

// ---------- helpers ----------
static __device__ __forceinline__ unsigned short bf_bits(float f) {
  union { float f; unsigned u; } v; v.f = f;
  return (unsigned short)((v.u + 0x7FFFu + ((v.u >> 16) & 1u)) >> 16);   // RNE
}

static __device__ __forceinline__ unsigned f2ord(float f) {
  unsigned u = __float_as_uint(f);
  return (u & 0x80000000u) ? ~u : (u | 0x80000000u);
}
static __device__ __forceinline__ float ord2f(unsigned u) {
  return (u & 0x80000000u) ? __uint_as_float(u & 0x7FFFFFFFu) : __uint_as_float(~u);
}
static __device__ __forceinline__ float lrelu(float x) {
  return x > 0.f ? x : NEG_SLOPE * x;
}
static __device__ __forceinline__ void edge_sd(const int* __restrict__ ei, int e,
                                               int& s, int& d) {
  if (e < N_EDGES) { s = ei[e]; d = ei[N_EDGES + e]; }
  else             { s = d = e - N_EDGES; }      // self-loop
}

// ---------- one-time f32 -> bf16 conversion ----------
__global__ void k_cvt_bf16(const float* __restrict__ src, ushortT* __restrict__ dst, int n) {
  int i = (blockIdx.x * blockDim.x + threadIdx.x) * 4;
  if (i >= n) return;                    // all sizes are multiples of 4
  float4 v = *(const float4*)(src + i);
  ushort4 o;
  o.x = bf_bits(v.x); o.y = bf_bits(v.y); o.z = bf_bits(v.z); o.w = bf_bits(v.w);
  *(ushort4*)(dst + i) = o;
}

// ---------- init scratch ----------
__global__ void k_init(unsigned* amax1, float* denom1, unsigned* amax2, float* denom2,
                       float* out1, float* out2) {
  long long t = (long long)blockIdx.x * blockDim.x + threadIdx.x;
  if (t < (long long)N_NODES * C2)   out2[t] = 0.f;
  if (t < (long long)N_NODES * HC1)  out1[t] = 0.f;
  if (t < (long long)N_NODES * HEADS1) { amax1[t] = ORD_NEG_INF; denom1[t] = 0.f; }
  if (t < N_NODES)                   { amax2[t] = ORD_NEG_INF; denom2[t] = 0.f; }
}

// ---------- WMMA GEMM on pre-converted bf16: C = A[M,K] @ W[Nout,K]^T ----------
// mode 0: fused linear-bias + eval BatchNorm, store bf16 to Cb
// mode 1: plain f32 store to Cf
__global__ __launch_bounds__(256)
void k_gemm_wmma(const ushortT* __restrict__ A, const ushortT* __restrict__ W,
                 float* __restrict__ Cf, ushortT* __restrict__ Cb, int K, int Nout,
                 const float* __restrict__ be, const float* __restrict__ bn_mean,
                 const float* __restrict__ bn_var, const float* __restrict__ bn_gamma,
                 const float* __restrict__ bn_beta, int mode) {
  const int wave = threadIdx.x >> 5;
  const int lane = threadIdx.x & 31;
  const int colTile = blockIdx.y * 8 + wave;
  if (colTile * 16 >= Nout) return;              // wave-uniform; WMMA waves keep EXEC all-1
  const int rowBase = blockIdx.x * 16;
  const int colBase = colTile * 16;
  const int rsel = lane & 15;                    // A row / W row (= output col)
  const int kOff = (lane >> 4) * 16;             // lane-group K offset
  const ushortT* ap = A + (size_t)(rowBase + rsel) * K + kOff;
  const ushortT* wp = W + (size_t)(colBase + rsel) * K + kOff;

  v8f acc = {};
  #pragma unroll 2                               // K is a multiple of 64
  for (int k0 = 0; k0 < K; k0 += 32) {
    v16bf af = *(const v16bf*)(ap + k0);         // 32B contiguous: 2x global_load_b128
    v16bf bfr = *(const v16bf*)(wp + k0);
    acc = __builtin_amdgcn_wmma_f32_16x16x32_bf16(false, af, false, bfr,
                                                  (short)0, acc, false, false);
  }

  const int row0 = rowBase + (lane >> 4) * 8;
  const int col  = colBase + (lane & 15);
  if (mode == 0) {
    float scale = rsqrtf(bn_var[col] + BN_EPS) * bn_gamma[col];
    float shift = (be[col] - bn_mean[col]) * scale + bn_beta[col];
    #pragma unroll
    for (int i = 0; i < 8; ++i)
      Cb[(size_t)(row0 + i) * Nout + col] = bf_bits(acc[i] * scale + shift);
  } else {
    #pragma unroll
    for (int i = 0; i < 8; ++i)
      Cf[(size_t)(row0 + i) * Nout + col] = acc[i];
  }
}

// ---------- attention coefficients ----------
__global__ void k_att1(const float* __restrict__ hf1, const float* __restrict__ as1,
                       const float* __restrict__ ad1, float* __restrict__ a_s,
                       float* __restrict__ a_d) {
  int t = blockIdx.x * blockDim.x + threadIdx.x;          // n*8 + h
  if (t >= N_NODES * HEADS1) return;
  int n = t >> 3, h = t & 7;
  const float* hp = hf1 + (size_t)n * HC1 + h * 8;
  float ss = 0.f, sd = 0.f;
  #pragma unroll
  for (int c = 0; c < 8; ++c) { ss += hp[c] * as1[h*8+c]; sd += hp[c] * ad1[h*8+c]; }
  a_s[t] = ss; a_d[t] = sd;
}

__global__ void k_att2(const float* __restrict__ hf2, const float* __restrict__ as2,
                       const float* __restrict__ ad2, float* __restrict__ a_s,
                       float* __restrict__ a_d) {
  int node = (blockIdx.x * blockDim.x + threadIdx.x) >> 5;
  int lane = threadIdx.x & 31;
  if (node >= N_NODES) return;
  const float* hp = hf2 + (size_t)node * C2 + lane * 8;
  float ss = 0.f, sd = 0.f;
  #pragma unroll
  for (int j = 0; j < 8; ++j) { float v = hp[j]; ss += v * as2[lane*8+j]; sd += v * ad2[lane*8+j]; }
  #pragma unroll
  for (int o = 16; o > 0; o >>= 1) { ss += __shfl_xor(ss, o, 32); sd += __shfl_xor(sd, o, 32); }
  if (lane == 0) { a_s[node] = ss; a_d[node] = sd; }
}

// ---------- edge passes, conv1 (8 heads) ----------
__global__ void e_max1(const int* __restrict__ ei, const float* __restrict__ a_s,
                       const float* __restrict__ a_d, unsigned* __restrict__ amax) {
  int t = blockIdx.x * blockDim.x + threadIdx.x;
  int e = t >> 3, h = t & 7;
  if (e >= NE_TOT) return;
  int s, d; edge_sd(ei, e, s, d);
  float al = lrelu(a_s[s*HEADS1+h] + a_d[d*HEADS1+h]);
  atomicMax(&amax[d*HEADS1+h], f2ord(al));
}

__global__ void e_sum1(const int* __restrict__ ei, const float* __restrict__ a_s,
                       const float* __restrict__ a_d, const unsigned* __restrict__ amax,
                       float* __restrict__ denom) {
  int t = blockIdx.x * blockDim.x + threadIdx.x;
  int e = t >> 3, h = t & 7;
  if (e >= NE_TOT) return;
  int s, d; edge_sd(ei, e, s, d);
  float al = lrelu(a_s[s*HEADS1+h] + a_d[d*HEADS1+h]);
  atomicAdd(&denom[d*HEADS1+h], __expf(al - ord2f(amax[d*HEADS1+h])));
}

__global__ void e_agg1(const int* __restrict__ ei, const float* __restrict__ a_s,
                       const float* __restrict__ a_d, const unsigned* __restrict__ amax,
                       const float* __restrict__ denom, const float* __restrict__ hf1,
                       float* __restrict__ out1) {
  int e = (blockIdx.x * blockDim.x + threadIdx.x) >> 5;   // one wave per edge
  int lane = threadIdx.x & 31;
  if (e >= NE_TOT) return;
  int s, d; edge_sd(ei, e, s, d);
  int h = lane >> 2;                                      // 2 features/lane, head = f/8
  float al = lrelu(a_s[s*HEADS1+h] + a_d[d*HEADS1+h]);
  float w  = __expf(al - ord2f(amax[d*HEADS1+h])) / (denom[d*HEADS1+h] + 1e-16f);
  float2 hv = *(const float2*)(hf1 + (size_t)s * HC1 + 2 * lane);
  atomicAdd(&out1[(size_t)d * HC1 + 2*lane    ], hv.x * w);
  atomicAdd(&out1[(size_t)d * HC1 + 2*lane + 1], hv.y * w);
}

__global__ void k_elu1(float* __restrict__ out1, const float* __restrict__ b1,
                       ushortT* __restrict__ h1b) {
  int t = blockIdx.x * blockDim.x + threadIdx.x;
  if (t >= N_NODES * HC1) return;
  float x = out1[t] + b1[t & (HC1 - 1)];
  x = x > 0.f ? x : (__expf(x) - 1.f);
  out1[t] = x;
  h1b[t] = bf_bits(x);                 // bf16 shadow feeding GEMM3
}

// ---------- edge passes, conv2 (1 head, 256 ch) ----------
__global__ void e_max2(const int* __restrict__ ei, const float* __restrict__ a_s,
                       const float* __restrict__ a_d, unsigned* __restrict__ amax) {
  int e = blockIdx.x * blockDim.x + threadIdx.x;
  if (e >= NE_TOT) return;
  int s, d; edge_sd(ei, e, s, d);
  atomicMax(&amax[d], f2ord(lrelu(a_s[s] + a_d[d])));
}

__global__ void e_sum2(const int* __restrict__ ei, const float* __restrict__ a_s,
                       const float* __restrict__ a_d, const unsigned* __restrict__ amax,
                       float* __restrict__ denom) {
  int e = blockIdx.x * blockDim.x + threadIdx.x;
  if (e >= NE_TOT) return;
  int s, d; edge_sd(ei, e, s, d);
  atomicAdd(&denom[d], __expf(lrelu(a_s[s] + a_d[d]) - ord2f(amax[d])));
}

__global__ void e_agg2(const int* __restrict__ ei, const float* __restrict__ a_s,
                       const float* __restrict__ a_d, const unsigned* __restrict__ amax,
                       const float* __restrict__ denom, const float* __restrict__ hf2,
                       float* __restrict__ out2) {
  int e = (blockIdx.x * blockDim.x + threadIdx.x) >> 5;   // one wave per edge
  int lane = threadIdx.x & 31;
  if (e >= NE_TOT) return;
  int s, d; edge_sd(ei, e, s, d);
  float w = __expf(lrelu(a_s[s] + a_d[d]) - ord2f(amax[d])) / (denom[d] + 1e-16f);
  const float4* hp = (const float4*)(hf2 + (size_t)s * C2 + lane * 8);
  float4 x0 = hp[0], x1 = hp[1];
  float* op = out2 + (size_t)d * C2 + lane * 8;
  atomicAdd(op + 0, x0.x * w); atomicAdd(op + 1, x0.y * w);
  atomicAdd(op + 2, x0.z * w); atomicAdd(op + 3, x0.w * w);
  atomicAdd(op + 4, x1.x * w); atomicAdd(op + 5, x1.y * w);
  atomicAdd(op + 6, x1.z * w); atomicAdd(op + 7, x1.w * w);
}

// ---------- final projection + zero second output ----------
__global__ void k_out(const float* __restrict__ out2, const float* __restrict__ b2,
                      const float* __restrict__ Wo, const float* __restrict__ bo,
                      float* __restrict__ dout) {
  int node = (blockIdx.x * blockDim.x + threadIdx.x) >> 5;
  int lane = threadIdx.x & 31;
  if (node >= N_NODES) return;
  const float* hp = out2 + (size_t)node * C2 + lane * 8;
  float s = 0.f;
  #pragma unroll
  for (int j = 0; j < 8; ++j) { int c = lane * 8 + j; s += (hp[j] + b2[c]) * Wo[c]; }
  #pragma unroll
  for (int o = 16; o > 0; o >>= 1) s += __shfl_xor(s, o, 32);
  if (lane == 0) { dout[node] = s + bo[0]; dout[N_NODES + node] = 0.f; }
}

extern "C" void kernel_launch(void* const* d_in, const int* in_sizes, int n_in,
                              void* d_out, int out_size, void* d_ws, size_t ws_size,
                              hipStream_t stream) {
  const float* x        = (const float*)d_in[0];
  const int*   ei       = (const int*)d_in[1];
  const float* We       = (const float*)d_in[2];
  const float* be       = (const float*)d_in[3];
  const float* bn_gamma = (const float*)d_in[4];
  const float* bn_beta  = (const float*)d_in[5];
  const float* bn_mean  = (const float*)d_in[6];
  const float* bn_var   = (const float*)d_in[7];
  const float* W1       = (const float*)d_in[8];
  const float* as1      = (const float*)d_in[9];
  const float* ad1      = (const float*)d_in[10];
  const float* b1       = (const float*)d_in[11];
  const float* W2       = (const float*)d_in[12];
  const float* as2      = (const float*)d_in[13];
  const float* ad2      = (const float*)d_in[14];
  const float* b2       = (const float*)d_in[15];
  const float* Wo       = (const float*)d_in[16];
  const float* bo       = (const float*)d_in[17];
  float* dout = (float*)d_out;

  // workspace carve-out
  char* w = (char*)d_ws;  size_t off = 0;
  ushortT* xb   = (ushortT*)(w + off); off += (size_t)N_NODES * F_IN * 2;
  ushortT* h0b  = (ushortT*)(w + off); off += (size_t)N_NODES * INT_D * 2;
  ushortT* h1b  = (ushortT*)(w + off); off += (size_t)N_NODES * HC1 * 2;
  ushortT* Web  = (ushortT*)(w + off); off += (size_t)INT_D * F_IN * 2;
  ushortT* W1b  = (ushortT*)(w + off); off += (size_t)HC1 * INT_D * 2;
  ushortT* W2b  = (ushortT*)(w + off); off += (size_t)C2 * HC1 * 2;
  off = (off + 255) & ~(size_t)255;
  float* hf1   = (float*)(w + off); off += (size_t)N_NODES * HC1 * 4;
  float* out1  = (float*)(w + off); off += (size_t)N_NODES * HC1 * 4;
  float* hf2   = (float*)(w + off); off += (size_t)N_NODES * C2 * 4;
  float* out2  = (float*)(w + off); off += (size_t)N_NODES * C2 * 4;
  float* a_s1  = (float*)(w + off); off += (size_t)N_NODES * HEADS1 * 4;
  float* a_d1  = (float*)(w + off); off += (size_t)N_NODES * HEADS1 * 4;
  unsigned* amax1 = (unsigned*)(w + off); off += (size_t)N_NODES * HEADS1 * 4;
  float* denom1   = (float*)(w + off);    off += (size_t)N_NODES * HEADS1 * 4;
  float* a_s2  = (float*)(w + off); off += (size_t)N_NODES * 4;
  float* a_d2  = (float*)(w + off); off += (size_t)N_NODES * 4;
  unsigned* amax2 = (unsigned*)(w + off); off += (size_t)N_NODES * 4;
  float* denom2   = (float*)(w + off);    off += (size_t)N_NODES * 4;

  const int TB = 256;
  // 0) init accumulators + one-time bf16 conversions
  k_init<<<(N_NODES * C2 + TB - 1) / TB, TB, 0, stream>>>(amax1, denom1, amax2, denom2, out1, out2);
  k_cvt_bf16<<<((N_NODES * F_IN / 4) + TB - 1) / TB, TB, 0, stream>>>(x, xb, N_NODES * F_IN);
  k_cvt_bf16<<<((INT_D * F_IN / 4) + TB - 1) / TB, TB, 0, stream>>>(We, Web, INT_D * F_IN);
  k_cvt_bf16<<<((HC1 * INT_D / 4) + TB - 1) / TB, TB, 0, stream>>>(W1, W1b, HC1 * INT_D);
  k_cvt_bf16<<<((C2 * HC1 / 4) + TB - 1) / TB, TB, 0, stream>>>(W2, W2b, C2 * HC1);
  // 1) h0 = BN(x @ We^T + be)   [WMMA bf16, bf16 output]
  k_gemm_wmma<<<dim3(N_NODES/16, 2), TB, 0, stream>>>(xb, Web, nullptr, h0b, F_IN, INT_D,
      be, bn_mean, bn_var, bn_gamma, bn_beta, 0);
  // 2) hf1 = h0 @ W1^T          [WMMA bf16, f32 output]
  k_gemm_wmma<<<dim3(N_NODES/16, 1), TB, 0, stream>>>(h0b, W1b, hf1, nullptr, INT_D, HC1,
      be, bn_mean, bn_var, bn_gamma, bn_beta, 1);
  // 3) per-node attention coefficients, conv1
  k_att1<<<(N_NODES * HEADS1 + TB - 1) / TB, TB, 0, stream>>>(hf1, as1, ad1, a_s1, a_d1);
  // 4) segment softmax + aggregate, conv1
  e_max1<<<((size_t)NE_TOT * 8 + TB - 1) / TB, TB, 0, stream>>>(ei, a_s1, a_d1, amax1);
  e_sum1<<<((size_t)NE_TOT * 8 + TB - 1) / TB, TB, 0, stream>>>(ei, a_s1, a_d1, amax1, denom1);
  e_agg1<<<((size_t)NE_TOT * 32 + TB - 1) / TB, TB, 0, stream>>>(ei, a_s1, a_d1, amax1, denom1, hf1, out1);
  // 5) bias + ELU (in place) + bf16 shadow
  k_elu1<<<(N_NODES * HC1 + TB - 1) / TB, TB, 0, stream>>>(out1, b1, h1b);
  // 6) hf2 = elu_out @ W2^T     [WMMA bf16, f32 output]
  k_gemm_wmma<<<dim3(N_NODES/16, 2), TB, 0, stream>>>(h1b, W2b, hf2, nullptr, HC1, C2,
      be, bn_mean, bn_var, bn_gamma, bn_beta, 1);
  // 7) attention coefficients, conv2
  k_att2<<<((size_t)N_NODES * 32 + TB - 1) / TB, TB, 0, stream>>>(hf2, as2, ad2, a_s2, a_d2);
  // 8) segment softmax + aggregate, conv2
  e_max2<<<(NE_TOT + TB - 1) / TB, TB, 0, stream>>>(ei, a_s2, a_d2, amax2);
  e_sum2<<<(NE_TOT + TB - 1) / TB, TB, 0, stream>>>(ei, a_s2, a_d2, amax2, denom2);
  e_agg2<<<((size_t)NE_TOT * 32 + TB - 1) / TB, TB, 0, stream>>>(ei, a_s2, a_d2, amax2, denom2, hf2, out2);
  // 9) final projection + zero sigma output
  k_out<<<((size_t)N_NODES * 32 + TB - 1) / TB, TB, 0, stream>>>(out2, b2, Wo, bo, dout);
}
